// myDecoder_ver_1_80796924773036
// MI455X (gfx1250) — matrix operand order repaired and checked by reading, not compiled, not run on hardware
//
#include <hip/hip_runtime.h>
#include <hip/hip_bf16.h>

// ---------- problem constants ----------
#define Bb   64
#define Ss   128
#define Hh   1024
#define Ee   1024
#define EMBD 512
#define Vv   32000
#define Tt   50
#define EOS_TOK 1

// ---------- WMMA types ----------
typedef __bf16 v16bf __attribute__((ext_vector_type(16)));
typedef float  v8f   __attribute__((ext_vector_type(8)));
typedef unsigned int u32x4 __attribute__((ext_vector_type(4)));

union BF16x16 { u32x4 u[2]; v16bf v; };

__device__ __forceinline__ __bf16 to_bf16(float f) { return (__bf16)f; }

// =====================================================================
// Register-blocked bf16 WMMA GEMM:  C[m, n] = sum_k A[m][k] * B[n][k]
//   A: [M][K] bf16 row-major (activations)
//   B: [N][K] bf16 row-major (weights in natural PyTorch layout)
//   C: f32, row m at C + m*ldc
// One wave computes a 32x64 macro-tile = 2(M) x 4(N) WMMA tiles.
// Per K-step: 2 A-fragments + 4 B-fragments (12 x b128 loads) feed
// 8 v_wmma_f32_16x16x32_bf16 -> ~21 FLOP per L2 byte, 8 independent
// accumulation chains for latency hiding.
// Requires M % 32 == 0, N % 64 == 0, K % 32 == 0 (true for all calls).
// =====================================================================
#define WMT 2
#define WNT 4
__global__ void wmma_gemm_bf16_kernel(const __bf16* __restrict__ A,
                                      const __bf16* __restrict__ B,
                                      float* __restrict__ C,
                                      int M, int N, int K, long long ldc)
{
    const int gwave = (blockIdx.x * blockDim.x + threadIdx.x) >> 5;
    const int lane  = threadIdx.x & 31;
    const int Mt = M >> 5;          // 32-row macro tiles
    const int Nt = N >> 6;          // 64-col macro tiles
    if (gwave >= Mt * Nt) return;   // uniform per wave: EXEC stays all-ones
    const int tm = gwave / Nt;
    const int tn = gwave - tm * Nt;
    const int l16  = lane & 15;
    const int half = lane >> 4;

    const __bf16* arow[WMT];
#pragma unroll
    for (int i = 0; i < WMT; ++i)
        arow[i] = A + (size_t)(tm * 32 + i * 16 + l16) * K;
    const __bf16* brow[WNT];
#pragma unroll
    for (int j = 0; j < WNT; ++j)
        brow[j] = B + (size_t)(tn * 64 + j * 16 + l16) * K;

    v8f acc[WMT][WNT];
#pragma unroll
    for (int i = 0; i < WMT; ++i)
#pragma unroll
        for (int j = 0; j < WNT; ++j)
            acc[i][j] = (v8f){0.f, 0.f, 0.f, 0.f, 0.f, 0.f, 0.f, 0.f};

    for (int k0 = 0; k0 < K; k0 += 32) {
        BF16x16 a[WMT], b[WNT];
#pragma unroll
        for (int i = 0; i < WMT; ++i) {
            // A fragment: lane (half,m) holds K = k0+half*16..+15 (contiguous)
            a[i].u[0] = *reinterpret_cast<const u32x4*>(arow[i] + k0 + half * 16);
            a[i].u[1] = *reinterpret_cast<const u32x4*>(arow[i] + k0 + half * 16 + 8);
        }
#pragma unroll
        for (int j = 0; j < WNT; ++j) {
            // B fragment: lane (half,n) holds K = k0+half*16..+15 (contiguous)
            const u32x4* bp = reinterpret_cast<const u32x4*>(brow[j] + k0 + half * 16);
            b[j].u[0] = bp[0];
            b[j].u[1] = bp[1];
        }
#pragma unroll
        for (int i = 0; i < WMT; ++i)
#pragma unroll
            for (int j = 0; j < WNT; ++j)
                acc[i][j] = __builtin_amdgcn_wmma_f32_16x16x32_bf16(
                    false, a[i].v, false, b[j].v, (short)0, acc[i][j], false, false);
    }

#pragma unroll
    for (int i = 0; i < WMT; ++i) {
        float* crow = C + (long long)(tm * 32 + i * 16 + half * 8) * ldc + tn * 64 + l16;
#pragma unroll
        for (int j = 0; j < WNT; ++j)
#pragma unroll
            for (int r = 0; r < 8; ++r)
                crow[(long long)r * ldc + j * 16] = acc[i][j][r];
    }
}

// =====================================================================
// Conversion / packing kernels
// =====================================================================
__global__ void conv_f32_bf16_kernel(const float* __restrict__ src,
                                     __bf16* __restrict__ dst, int n)
{
    int i = blockIdx.x * blockDim.x + threadIdx.x;
    if (i < n) dst[i] = to_bf16(src[i]);
}

// W_cat[j][k] = k<512 ? W_ih[j][k] : W_hh[j][k-512]   -> [4096][1536] bf16
__global__ void build_wcat_kernel(const float* __restrict__ Wih,
                                  const float* __restrict__ Whh,
                                  __bf16* __restrict__ Wcat)
{
    int i = blockIdx.x * blockDim.x + threadIdx.x;
    if (i >= 4 * Hh * (EMBD + Hh)) return;
    int j = i / (EMBD + Hh);
    int k = i - j * (EMBD + Hh);
    float v = (k < EMBD) ? Wih[(size_t)j * EMBD + k]
                         : Whh[(size_t)j * Hh + (k - EMBD)];
    Wcat[i] = to_bf16(v);
}

// =====================================================================
// Init: h=dec_init[0,0], c=dec_init[1,0]; xcat prev-part = 0,
// xcat h-part = bf16(h0); tl=fl=0
// =====================================================================
__global__ void init_kernel(const float* __restrict__ dec_init,
                            float* __restrict__ h, float* __restrict__ c,
                            __bf16* __restrict__ xcat,
                            int* __restrict__ tl, int* __restrict__ fl)
{
    int i = blockIdx.x * blockDim.x + threadIdx.x;    // Bb*Hh
    if (i >= Bb * Hh) return;
    int b = i >> 10, j = i & (Hh - 1);
    float h0 = dec_init[i];
    float c0 = dec_init[Bb * Hh + i];
    h[i] = h0;
    c[i] = c0;
    xcat[(size_t)b * (EMBD + Hh) + EMBD + j] = to_bf16(h0);
    if (j < EMBD) xcat[(size_t)b * (EMBD + Hh) + j] = to_bf16(0.f);
    if (i < Bb) { tl[i] = 0; fl[i] = 0; }
}

// =====================================================================
// LSTM elementwise: gates [B][4H] (i,f,g,o) + bias -> update c, h;
// writes bf16 h into xcat (next-step gates GEMM A) and hc_cat
// (out-proj GEMM A)
// =====================================================================
__global__ void lstm_kernel(const float* __restrict__ gates,
                            const float* __restrict__ b_lstm,
                            float* __restrict__ h, float* __restrict__ c,
                            __bf16* __restrict__ xcat,
                            __bf16* __restrict__ hc_cat)
{
    int i = blockIdx.x * blockDim.x + threadIdx.x;    // Bb*Hh
    if (i >= Bb * Hh) return;
    int b = i >> 10, j = i & (Hh - 1);
    const float* g = gates + (size_t)b * 4 * Hh;
    float gi = g[j]          + b_lstm[j];
    float gf = g[Hh + j]     + b_lstm[Hh + j];
    float gg = g[2 * Hh + j] + b_lstm[2 * Hh + j];
    float go = g[3 * Hh + j] + b_lstm[3 * Hh + j];
    float si = 1.f / (1.f + __expf(-gi));
    float sf = 1.f / (1.f + __expf(-gf));
    float so = 1.f / (1.f + __expf(-go));
    float cn = sf * c[i] + si * tanhf(gg);
    float hn = so * tanhf(cn);
    c[i] = cn;
    h[i] = hn;
    xcat[(size_t)b * (EMBD + Hh) + EMBD + j] = to_bf16(hn);
    hc_cat[(size_t)b * 2 * Hh + j] = to_bf16(hn);
}

// =====================================================================
// Attention: one block per batch row.
//  scores[s] = keys[b,s,:].h[b,:]  (8 waves x 16 rows, wave-reduced)
//  masked softmax over S in LDS
//  c_t[e] = sum_s alpha[s]*enc[b,s,e]; also bf16 into hc_cat[:,1024+e]
// =====================================================================
__global__ void attn_kernel(const float* __restrict__ keys,
                            const float* __restrict__ h,
                            const float* __restrict__ enc,
                            const int* __restrict__ enc_len,
                            float* __restrict__ c_t,
                            __bf16* __restrict__ hc_cat)
{
    const int b   = blockIdx.x;
    const int tid = threadIdx.x;          // 256 threads = 8 waves
    const int wv  = tid >> 5, ln = tid & 31;
    __shared__ float sc[Ss];
    __shared__ float red[256];

    const float* hr = h + (size_t)b * Hh;
    const int    L  = enc_len[b];

    for (int s = wv; s < Ss; s += 8) {
        const float* kr = keys + ((size_t)b * Ss + s) * Hh;
        float acc = 0.f;
        for (int e = ln; e < Hh; e += 32) acc += kr[e] * hr[e];
#pragma unroll
        for (int off = 16; off > 0; off >>= 1) acc += __shfl_down(acc, off, 32);
        if (ln == 0) sc[s] = (s < L) ? acc : -1e9f;
    }
    __syncthreads();

    // max
    red[tid] = (tid < Ss) ? sc[tid] : -1e9f;
    __syncthreads();
    for (int off = 128; off >= 1; off >>= 1) {
        if (tid < off) red[tid] = fmaxf(red[tid], red[tid + off]);
        __syncthreads();
    }
    float mx = red[0];
    __syncthreads();
    // sum of exp
    float ex = (tid < Ss) ? __expf(sc[tid] - mx) : 0.f;
    red[tid] = ex;
    __syncthreads();
    for (int off = 128; off >= 1; off >>= 1) {
        if (tid < off) red[tid] += red[tid + off];
        __syncthreads();
    }
    float inv = 1.f / red[0];
    __syncthreads();
    if (tid < Ss) sc[tid] = ex * inv;
    __syncthreads();

    // context
    for (int e = tid; e < Ee; e += 256) {
        const float* ecol = enc + ((size_t)b * Ss) * Ee + e;
        float acc = 0.f;
        for (int s = 0; s < Ss; ++s) acc += sc[s] * ecol[(size_t)s * Ee];
        c_t[(size_t)b * Ee + e] = acc;
        hc_cat[(size_t)b * 2 * Hh + Hh + e] = to_bf16(acc);
    }
}

// =====================================================================
// Per-step vocab row scan: argmax (first-max tie rule), row max, log-sum-exp.
// Logits already live in d_out[b][t][:].
// =====================================================================
__global__ void vocab_reduce_kernel(const float* __restrict__ out, int t,
                                    int* __restrict__ sym,
                                    float* __restrict__ rowmax,
                                    float* __restrict__ rowlse)
{
    const int b = blockIdx.x, tid = threadIdx.x;   // 256 threads
    const float* row = out + ((size_t)b * Tt + t) * Vv;
    __shared__ float smax[256];
    __shared__ int   sarg[256];
    __shared__ float ssum[256];

    float mx = -3.4e38f; int arg = 0;
    for (int v = tid; v < Vv; v += 256) {
        float x = row[v];
        if (x > mx) { mx = x; arg = v; }
    }
    smax[tid] = mx; sarg[tid] = arg;
    __syncthreads();
    for (int off = 128; off >= 1; off >>= 1) {
        if (tid < off) {
            float mo = smax[tid + off]; int ao = sarg[tid + off];
            if (mo > smax[tid] || (mo == smax[tid] && ao < sarg[tid])) {
                smax[tid] = mo; sarg[tid] = ao;
            }
        }
        __syncthreads();
    }
    float m = smax[0];
    __syncthreads();

    float se = 0.f;
    for (int v = tid; v < Vv; v += 256) se += __expf(row[v] - m);
    ssum[tid] = se;
    __syncthreads();
    for (int off = 128; off >= 1; off >>= 1) {
        if (tid < off) ssum[tid] += ssum[tid + off];
        __syncthreads();
    }
    if (tid == 0) {
        sym[b] = sarg[0];
        rowmax[b * Tt + t] = m;
        rowlse[b * Tt + t] = __logf(ssum[0]);
    }
}

// =====================================================================
// Update: prev = emb_table[sym]; xcat prev-part = bf16(prev); tl/fl logic
// =====================================================================
__global__ void update_kernel(const int* __restrict__ sym,
                              const float* __restrict__ emb,
                              __bf16* __restrict__ xcat,
                              int* __restrict__ tl, int* __restrict__ fl, int t)
{
    int i = blockIdx.x * blockDim.x + threadIdx.x;   // Bb*EMBD
    if (i >= Bb * EMBD) return;
    int b = i >> 9, j = i & (EMBD - 1);
    int s = sym[b];
    float e = emb[(size_t)s * EMBD + j];
    xcat[(size_t)b * (EMBD + Hh) + j] = to_bf16(e);
    if (j == 0) {
        bool is_eos = (s == EOS_TOK);
        if (is_eos) { tl[b] = t + 1; fl[b] = 1; }
        if (t == Tt - 1 && !is_eos) tl[b] = Tt;    // tl = where(~eos_hist[-1], T, tl)
    }
}

// =====================================================================
// In-place log_softmax over V using precomputed row max / lse
// =====================================================================
__global__ void logsoftmax_kernel(float* __restrict__ out,
                                  const float* __restrict__ rowmax,
                                  const float* __restrict__ rowlse)
{
    size_t i = (size_t)blockIdx.x * blockDim.x + threadIdx.x;
    const size_t total = (size_t)Bb * Tt * Vv;
    if (i >= total) return;
    size_t bt = i / Vv;
    out[i] = out[i] - rowmax[bt] - rowlse[bt];
}

__global__ void write_tl_kernel(const int* __restrict__ tl, float* __restrict__ out)
{
    int b = threadIdx.x;
    if (b < Bb) reinterpret_cast<int*>(out + (size_t)Bb * Tt * Vv)[b] = tl[b];
}

// =====================================================================
// Host launcher
// =====================================================================
extern "C" void kernel_launch(void* const* d_in, const int* in_sizes, int n_in,
                              void* d_out, int out_size, void* d_ws, size_t ws_size,
                              hipStream_t stream)
{
    const float* dec_init = (const float*)d_in[0];
    const float* enc      = (const float*)d_in[1];
    const int*   enc_len  = (const int*)  d_in[2];
    const float* Wih      = (const float*)d_in[4];
    const float* Whh      = (const float*)d_in[5];
    const float* b_lstm   = (const float*)d_in[6];
    const float* Watt     = (const float*)d_in[7];
    const float* Wout     = (const float*)d_in[8];
    const float* Wvoc     = (const float*)d_in[9];
    const float* emb      = (const float*)d_in[10];
    float* out = (float*)d_out;

    // ---- workspace carve (256B aligned) ----
    char* p = (char*)d_ws;
    auto carve = [&](size_t bytes) {
        char* r = p;
        p += (bytes + 255) & ~(size_t)255;
        return r;
    };
    __bf16* wcat_bf  = (__bf16*)carve((size_t)4 * Hh * (EMBD + Hh) * 2); // [4096][1536]
    __bf16* watt_bf  = (__bf16*)carve((size_t)Hh * Ee * 2);              // [1024][1024]
    __bf16* wout_bf  = (__bf16*)carve((size_t)EMBD * 2 * Hh * 2);        // [512][2048]
    __bf16* wvoc_bf  = (__bf16*)carve((size_t)Vv * EMBD * 2);            // [32000][512]
    __bf16* enc_bf   = (__bf16*)carve((size_t)Bb * Ss * Ee * 2);         // [8192][1024]
    float*  keys     = (float*) carve((size_t)Bb * Ss * Hh * 4);         // [8192][1024]
    float*  hbuf     = (float*) carve((size_t)Bb * Hh * 4);
    float*  cbuf     = (float*) carve((size_t)Bb * Hh * 4);
    __bf16* xcat     = (__bf16*)carve((size_t)Bb * (EMBD + Hh) * 2);     // [64][1536]
    float*  gates    = (float*) carve((size_t)Bb * 4 * Hh * 4);          // [64][4096]
    __bf16* hc_cat   = (__bf16*)carve((size_t)Bb * 2 * Hh * 2);          // [64][2048]
    float*  c_t      = (float*) carve((size_t)Bb * Ee * 4);
    float*  dec      = (float*) carve((size_t)Bb * EMBD * 4);
    __bf16* dec_bf   = (__bf16*)carve((size_t)Bb * EMBD * 2);
    float*  rowmax   = (float*) carve((size_t)Bb * Tt * 4);
    float*  rowlse   = (float*) carve((size_t)Bb * Tt * 4);
    int*    sym      = (int*)   carve(Bb * 4);
    int*    tl       = (int*)   carve(Bb * 4);
    int*    fl       = (int*)   carve(Bb * 4);
    (void)in_sizes; (void)n_in; (void)out_size; (void)ws_size;

    const int TB = 256;
    auto blocks = [](long long n, int tb) { return (unsigned)((n + tb - 1) / tb); };
    auto gemm_blocks = [&](int M, int N) {
        long long waves = (long long)(M >> 5) * (N >> 6);   // 32x64 macro tiles
        return (unsigned)((waves * 32 + TB - 1) / TB);
    };

    // ---- one-time (per launch) conversions ----
    build_wcat_kernel<<<blocks((long long)4 * Hh * (EMBD + Hh), TB), TB, 0, stream>>>(Wih, Whh, wcat_bf);
    conv_f32_bf16_kernel<<<blocks((long long)Hh * Ee, TB), TB, 0, stream>>>(Watt, watt_bf, Hh * Ee);
    conv_f32_bf16_kernel<<<blocks((long long)EMBD * 2 * Hh, TB), TB, 0, stream>>>(Wout, wout_bf, EMBD * 2 * Hh);
    conv_f32_bf16_kernel<<<blocks((long long)Vv * EMBD, TB), TB, 0, stream>>>(Wvoc, wvoc_bf, Vv * EMBD);
    conv_f32_bf16_kernel<<<blocks((long long)Bb * Ss * Ee, TB), TB, 0, stream>>>(enc, enc_bf, Bb * Ss * Ee);

    // ---- keys = enc @ W_attproj^T : [8192][1024] x [1024][1024] ----
    wmma_gemm_bf16_kernel<<<gemm_blocks(Bb * Ss, Hh), TB, 0, stream>>>(
        enc_bf, watt_bf, keys, Bb * Ss, Hh, Ee, (long long)Hh);

    init_kernel<<<blocks(Bb * Hh, TB), TB, 0, stream>>>(dec_init, hbuf, cbuf, xcat, tl, fl);

    // ---- sequential decode ----
    for (int t = 0; t < Tt; ++t) {
        // gates = xcat @ Wcat^T : [64][1536] x [4096][1536]
        wmma_gemm_bf16_kernel<<<gemm_blocks(Bb, 4 * Hh), TB, 0, stream>>>(
            xcat, wcat_bf, gates, Bb, 4 * Hh, EMBD + Hh, (long long)(4 * Hh));
        lstm_kernel<<<blocks(Bb * Hh, TB), TB, 0, stream>>>(gates, b_lstm, hbuf, cbuf, xcat, hc_cat);
        attn_kernel<<<Bb, TB, 0, stream>>>(keys, hbuf, enc, enc_len, c_t, hc_cat);
        // dec = [h, c_t] @ W_out^T : [64][2048] x [512][2048]
        wmma_gemm_bf16_kernel<<<gemm_blocks(Bb, EMBD), TB, 0, stream>>>(
            hc_cat, wout_bf, dec, Bb, EMBD, 2 * Hh, (long long)EMBD);
        conv_f32_bf16_kernel<<<blocks(Bb * EMBD, TB), TB, 0, stream>>>(dec, dec_bf, Bb * EMBD);
        // logits -> d_out[b][t][:] : [64][512] x [32000][512], row stride T*V
        wmma_gemm_bf16_kernel<<<gemm_blocks(Bb, Vv), TB, 0, stream>>>(
            dec_bf, wvoc_bf, out + (size_t)t * Vv, Bb, Vv, EMBD, (long long)Tt * Vv);
        vocab_reduce_kernel<<<Bb, TB, 0, stream>>>(out, t, sym, rowmax, rowlse);
        update_kernel<<<blocks(Bb * EMBD, TB), TB, 0, stream>>>(sym, emb, xcat, tl, fl, t);
    }

    // ---- finalize: in-place log_softmax + tl tail ----
    logsoftmax_kernel<<<blocks((long long)Bb * Tt * Vv, TB), TB, 0, stream>>>(out, rowmax, rowlse);
    write_tl_kernel<<<1, 64, 0, stream>>>(tl, out);
}
